// UnionEdgeModel_81372450390358
// MI455X (gfx1250) — matrix operand order repaired and checked by reading, not compiled, not run on hardware
//
#include <hip/hip_runtime.h>
#include <stdint.h>
#include <stddef.h>

// ---------------------------------------------------------------------------
// UnionEdgeModel on MI455X (gfx1250).
// Roofline: 4 SpMMs over 1M edges x 128 fp32 features dominate (~4 GB of
// gather + atomic-scatter traffic -> ~190us floor @ 23.3 TB/s). The 6 dense
// 65536x128x128 GEMMs (~13 GFLOP) go through v_wmma_f32_16x16x32_bf16.
// GEMM A-operand: one workgroup per 16-row tile; block converts the fp32 tile
// to bf16 in LDS once (native __bf16 cvt), 8 waves (= 8 ntiles) then build
// fragments with ds_read_b128 from the padded LDS tile -> minimal VALU.
// Pooling into 64 groups accumulates in LDS (ds_add_f32) before one global
// flush per block, avoiding L2 atomic serialization.
// ---------------------------------------------------------------------------

typedef __bf16 bf16_t;
typedef bf16_t v16bf __attribute__((ext_vector_type(16)));
typedef float  v8f   __attribute__((ext_vector_type(8)));

#define H_DIM   128
#define ALPHA_F 0.9f
#define BGROUPS 64
#define LDS_STRIDE 68   // 64 payload dwords + 4 pad: bank = (4*row + c) % 64

union Frag256 { uint32_t u[8]; v16bf v; };

__device__ __forceinline__ uint32_t pack_bf16x2(float lo, float hi) {
  union { bf16_t h[2]; uint32_t u; } cv;
  cv.h[0] = (bf16_t)lo;        // native fptrunc -> v_cvt bf16 (RNE)
  cv.h[1] = (bf16_t)hi;
  return cv.u;
}

// ---------------------------------------------------------------------------
// Repack a 128x128 row-major fp32 weight into the WMMA B-operand lane-striped
// bf16 layout: per (ktile, ntile) a 32x16 (KxN) tile; lane l (0..31) holds
// K = kt*32 + (l>=16 ? 16:0) + 2j .. +1 at column n = ntile*16 + (l&15),
// two bf16 per dword, 8 dwords per lane, fragments stored contiguously.
// 6 weights in one launch (t in [0, 6*8192) output dwords).
// ---------------------------------------------------------------------------
__global__ void pack_weights_kernel(const float* w0, const float* w1,
                                    const float* w2, const float* w3,
                                    const float* w4, const float* w5,
                                    uint32_t* __restrict__ dst) {
  int t = blockIdx.x * blockDim.x + threadIdx.x;
  if (t >= 6 * 8192) return;
  int wi = t >> 13;
  int r  = t & 8191;
  const float* W = (wi == 0) ? w0 : (wi == 1) ? w1 : (wi == 2) ? w2
                 : (wi == 3) ? w3 : (wi == 4) ? w4 : w5;
  int j     = r & 7;
  int lane  = (r >> 3) & 31;
  int ntile = (r >> 8) & 7;
  int kt    = r >> 11;
  int k0 = kt * 32 + ((lane >> 4) << 4) + (j << 1);
  int n  = (ntile << 4) + (lane & 15);
  dst[t] = pack_bf16x2(W[k0 * H_DIM + n], W[(k0 + 1) * H_DIM + n]);
}

// ---------------------------------------------------------------------------
// Out[nRows,128] = relu(A[nRows,128] @ W + bias), W pre-packed bf16.
// Block = 8 waves = 8 ntiles of one 16-row tile. Stage A tile fp32->bf16 in
// LDS once, then per wave: 4 x (two ds_read_b128 A-frag + two b128 B-frag +
// v_wmma_f32_16x16x32_bf16), bias+relu, scatter fp32 stores.
// nRows must be a multiple of 16 (grid = nRows/16 blocks).
// ---------------------------------------------------------------------------
__global__ void __launch_bounds__(256)
gemm_bias_relu_kernel(const float* __restrict__ A,
                      const uint32_t* __restrict__ Wpk,
                      const float* __restrict__ bias,
                      float* __restrict__ Out) {
  __shared__ uint32_t atile[16 * LDS_STRIDE];   // 16 rows x 64 bf16-pair dwords
  const int tileM = blockIdx.x;

  // ---- cooperative stage: 16x128 fp32 -> bf16 LDS (each element once) ----
  {
    int t   = threadIdx.x;
    int row = t >> 4;                 // 16 threads per row
    int c4  = (t & 15) << 2;          // dword offset within row (0,4,...,60)
    const float* src = A + ((size_t)tileM * 16 + row) * H_DIM + (c4 << 1);
    float4 f01 = *(const float4*)(src);
    float4 f23 = *(const float4*)(src + 4);
    uint4 pk;
    pk.x = pack_bf16x2(f01.x, f01.y);
    pk.y = pack_bf16x2(f01.z, f01.w);
    pk.z = pack_bf16x2(f23.x, f23.y);
    pk.w = pack_bf16x2(f23.z, f23.w);
    *(uint4*)&atile[row * LDS_STRIDE + c4] = pk;
  }
  __syncthreads();

  const int ntile = threadIdx.x >> 5;           // wave id = output col tile
  const int lane  = threadIdx.x & 31;
  const int half  = lane >> 4;
  const int lmod  = lane & 15;

  v8f acc = {};
#pragma unroll
  for (int kt = 0; kt < 4; ++kt) {
    Frag256 a, b;
    // A striping: vgpr j holds bf16 pair K = kt*32 + half*16 + 2*j
    // -> dword index kbase/2 = kt*16 + half*8 (+4 for upper four vgprs)
    const uint32_t* ap = &atile[lmod * LDS_STRIDE + kt * 16 + half * 8];
    *(uint4*)&a.u[0] = *(const uint4*)(ap);
    *(uint4*)&a.u[4] = *(const uint4*)(ap + 4);
    const uint32_t* bp = Wpk + (size_t)((((kt << 3) + ntile) << 5) + lane) * 8;
    *(uint4*)&b.u[0] = *(const uint4*)(bp);
    *(uint4*)&b.u[4] = *(const uint4*)(bp + 4);
    acc = __builtin_amdgcn_wmma_f32_16x16x32_bf16(
        /*neg_a=*/false, a.v, /*neg_b=*/false, b.v,
        /*c_mod=*/(short)0, acc, /*reuse_a=*/false, /*reuse_b=*/false);
  }

  const int n  = (ntile << 4) + lmod;
  const float bs = bias[n];
#pragma unroll
  for (int r = 0; r < 8; ++r) {
    // C/D layout: vgpr r -> row M = r + (lane>=16 ? 8:0), col N = lane&15
    int m = tileM * 16 + r + half * 8;
    float v = acc[r] + bs;
    Out[(size_t)m * H_DIM + n] = v > 0.f ? v : 0.f;
  }
}

// ---------------------------------------------------------------------------
// agg[dst] += w * X[src] over all edges; one wave per edge, 32 lanes x float4
// cover the 128-wide row. Coalesced gather, global_atomic_add_f32 scatter.
// edge_index is (2,E) row-major int32.
// ---------------------------------------------------------------------------
__global__ void __launch_bounds__(256)
spmm_scatter_kernel(const float* __restrict__ X, const int* __restrict__ ei,
                    const float* __restrict__ ew, float* __restrict__ agg,
                    int nEdges) {
  int wave = (blockIdx.x * blockDim.x + threadIdx.x) >> 5;
  int lane = threadIdx.x & 31;
  if (wave >= nEdges) return;
  int   src = ei[wave];
  int   dst = ei[nEdges + wave];
  float wt  = ew[wave];
  float4 v = ((const float4*)(X + (size_t)src * H_DIM))[lane];
  float* d = agg + (size_t)dst * H_DIM + lane * 4;
  atomicAdd(d + 0, v.x * wt);
  atomicAdd(d + 1, v.y * wt);
  atomicAdd(d + 2, v.z * wt);
  atomicAdd(d + 3, v.w * wt);
}

// agg[m] = ALPHA*agg[m] + (1-ALPHA)*base_x[copy2orig[m]]  (in place, per row)
__global__ void __launch_bounds__(256)
mix_kernel(float* __restrict__ agg, const float* __restrict__ base_x,
           const int* __restrict__ copy2orig, int mRows) {
  int wave = (blockIdx.x * blockDim.x + threadIdx.x) >> 5;
  int lane = threadIdx.x & 31;
  if (wave >= mRows) return;
  int o = copy2orig[wave];
  float4 b = ((const float4*)(base_x + (size_t)o * H_DIM))[lane];
  float4* ap = (float4*)(agg + (size_t)wave * H_DIM) + lane;
  float4 a = *ap;
  a.x = ALPHA_F * a.x + (1.f - ALPHA_F) * b.x;
  a.y = ALPHA_F * a.y + (1.f - ALPHA_F) * b.y;
  a.z = ALPHA_F * a.z + (1.f - ALPHA_F) * b.z;
  a.w = ALPHA_F * a.w + (1.f - ALPHA_F) * b.w;
  *ap = a;
}

__global__ void zero_kernel(float4* p, size_t n4) {
  size_t t = blockIdx.x * (size_t)blockDim.x + threadIdx.x;
  if (t < n4) p[t] = make_float4(0.f, 0.f, 0.f, 0.f);
}

// ---------------------------------------------------------------------------
// Segment-sum pooling of K entries into 64 groups x 128 features.
// Per-block LDS accumulator (64*128 fp32 = 32KB of the 320KB WGP LDS) with
// ds_add_f32 atomics, then one global atomic flush per block.
// ---------------------------------------------------------------------------
__global__ void __launch_bounds__(256)
pool_kernel(const float* __restrict__ local_x, const int* __restrict__ midx,
            const int* __restrict__ mgid, float* __restrict__ pooled,
            float* __restrict__ cnt, int K) {
  __shared__ float acc[BGROUPS * H_DIM];
  __shared__ float c[BGROUPS];
  for (int i = threadIdx.x; i < BGROUPS * H_DIM; i += blockDim.x) acc[i] = 0.f;
  for (int i = threadIdx.x; i < BGROUPS; i += blockDim.x) c[i] = 0.f;
  __syncthreads();

  int wavesPerBlock = blockDim.x >> 5;
  int wave = threadIdx.x >> 5;
  int lane = threadIdx.x & 31;
  int per  = (K + gridDim.x - 1) / gridDim.x;
  int s = blockIdx.x * per;
  int e = s + per; if (e > K) e = K;
  for (int k = s + wave; k < e; k += wavesPerBlock) {
    int idx = midx[k];
    int gid = mgid[k] & (BGROUPS - 1);
    float4 v = ((const float4*)(local_x + (size_t)idx * H_DIM))[lane];
    float* a = acc + gid * H_DIM + lane * 4;
    atomicAdd(a + 0, v.x);
    atomicAdd(a + 1, v.y);
    atomicAdd(a + 2, v.z);
    atomicAdd(a + 3, v.w);
    if (lane == 0) atomicAdd(&c[gid], 1.f);
  }
  __syncthreads();
  for (int i = threadIdx.x; i < BGROUPS * H_DIM; i += blockDim.x)
    atomicAdd(&pooled[i], acc[i]);
  for (int i = threadIdx.x; i < BGROUPS; i += blockDim.x)
    atomicAdd(&cnt[i], c[i]);
}

// out[g, o] = (pooled[g]/max(cnt[g],1)) . Wp[:, o] + bp[o]   (64x128 @ 128x2)
__global__ void head_kernel(const float* __restrict__ pooled,
                            const float* __restrict__ cnt,
                            const float* __restrict__ Wp,
                            const float* __restrict__ bpv,
                            float* __restrict__ out) {
  int t = threadIdx.x;                 // 128 threads: t = g*2 + o
  if (t >= BGROUPS * 2) return;
  int g = t >> 1, o = t & 1;
  float cc = cnt[g]; cc = cc > 1.f ? cc : 1.f;
  float s = 0.f;
  for (int k = 0; k < H_DIM; ++k) s += pooled[g * H_DIM + k] * Wp[k * 2 + o];
  out[t] = s / cc + bpv[o];
}

// ---------------------------------------------------------------------------
extern "C" void kernel_launch(void* const* d_in, const int* in_sizes, int n_in,
                              void* d_out, int out_size, void* d_ws, size_t ws_size,
                              hipStream_t stream) {
  const float* x           = (const float*)d_in[0];
  const int*   edge_index  = (const int*)  d_in[1];
  const float* edge_weight = (const float*)d_in[2];
  const float* local_x0    = (const float*)d_in[3];
  const int*   copy2orig   = (const int*)  d_in[4];
  const int*   ladj_idx    = (const int*)  d_in[5];
  const float* ladj_val    = (const float*)d_in[6];
  const int*   memb_idx    = (const int*)  d_in[7];
  const int*   memb_gid    = (const int*)  d_in[8];
  // d_in[9] = B (compile-time 64 here)
  const float* Wb_in   = (const float*)d_in[10];
  const float* bb_in   = (const float*)d_in[11];
  const float* Wl_in   = (const float*)d_in[12];
  const float* bl_in   = (const float*)d_in[13];
  const float* W_base  = (const float*)d_in[14];
  const float* b_base  = (const float*)d_in[15];
  const float* W_local = (const float*)d_in[16];
  const float* b_local = (const float*)d_in[17];
  const float* Wp      = (const float*)d_in[18];
  const float* bpv     = (const float*)d_in[19];

  const int N  = in_sizes[0] / H_DIM;   // 65536
  const int M  = in_sizes[3] / H_DIM;   // 65536
  const int E  = in_sizes[2];           // 1048576
  const int EL = in_sizes[6];           // 1048576
  const int K  = in_sizes[7];           // 131072
  const int L  = in_sizes[15] / H_DIM;  // 2

  // ---- workspace carve (bytes) ----
  char* w = (char*)d_ws;
  uint32_t* wpk     = (uint32_t*)w;                      // 6*8192 dwords = 192KB
  float*    base_x  = (float*)(w + 6 * 8192 * 4);        // N*128 f32 (32MB)
  float*    local_x = base_x  + (size_t)N * H_DIM;       // M*128 f32 (32MB)
  float*    agg     = local_x + (size_t)M * H_DIM;       // M*128 f32 (32MB)
  float*    pooled  = agg     + (size_t)M * H_DIM;       // 64*128 f32
  float*    cnt     = pooled  + BGROUPS * H_DIM;         // 64 f32

  // packed weight slots: 0=Wb_in 1=Wl_in 2..3=W_base[l] 4..5=W_local[l]
  pack_weights_kernel<<<(6 * 8192 + 255) / 256, 256, 0, stream>>>(
      Wb_in, Wl_in, W_base, W_base + 16384, W_local, W_local + 16384, wpk);

  const int gemmBlocksN = N / 16;   // one block per 16-row tile
  const int gemmBlocksM = M / 16;

  // input projections
  gemm_bias_relu_kernel<<<gemmBlocksN, 256, 0, stream>>>(x, wpk, bb_in, base_x);
  gemm_bias_relu_kernel<<<gemmBlocksM, 256, 0, stream>>>(local_x0, wpk + 8192, bl_in, local_x);

  const size_t agg4  = (size_t)M * H_DIM / 4;
  const int zeroBlks = (int)((agg4 + 255) / 256);

  for (int l = 0; l < L; ++l) {
    // base graph: agg = A_norm @ base_x ; base_x = relu(agg @ Wb[l] + bb[l])
    zero_kernel<<<zeroBlks, 256, 0, stream>>>((float4*)agg, agg4);
    spmm_scatter_kernel<<<(E * 32) / 256, 256, 0, stream>>>(base_x, edge_index, edge_weight, agg, E);
    gemm_bias_relu_kernel<<<gemmBlocksN, 256, 0, stream>>>(
        agg, wpk + (2 + l) * 8192, b_base + l * H_DIM, base_x);

    // local graph: agg = A_loc @ local_x ; mix with base via copy2orig ; GEMM
    zero_kernel<<<zeroBlks, 256, 0, stream>>>((float4*)agg, agg4);
    spmm_scatter_kernel<<<(EL * 32) / 256, 256, 0, stream>>>(local_x, ladj_idx, ladj_val, agg, EL);
    mix_kernel<<<(M * 32) / 256, 256, 0, stream>>>(agg, base_x, copy2orig, M);
    gemm_bias_relu_kernel<<<gemmBlocksM, 256, 0, stream>>>(
        agg, wpk + (4 + l) * 8192, b_local + l * H_DIM, local_x);
  }

  // pooling (sum + count) then head
  const size_t pool4 = (BGROUPS * H_DIM + BGROUPS) / 4;
  zero_kernel<<<(int)((pool4 + 255) / 256), 256, 0, stream>>>((float4*)pooled, pool4);
  pool_kernel<<<256, 256, 0, stream>>>(local_x, memb_idx, memb_gid, pooled, cnt, K);
  head_kernel<<<1, 128, 0, stream>>>(pooled, cnt, Wp, bpv, (float*)d_out);
}
